// SS_Linear_10574209483234
// MI455X (gfx1250) — matrix operand order repaired and checked by reading, not compiled, not run on hardware
//
#include <hip/hip_runtime.h>
#include <hip/hip_bf16.h>

#define NUM_CLASSES 100000
#define NUM_SAMPLED 5000
#define BATCH 4096
#define DIM 1024
#define SPAD 5120              // NUM_SAMPLED padded to multiple of 256
#define NCHUNK (SPAD / 256)    // 20 column chunks

// GEMM tiling
#define KCH 32                 // K elements per staged chunk (one WMMA K-step)
#define APITCH 40              // LDS row pitch in elements (32 + 8 pad -> conflict-free)
#define A_EL (64 * APITCH)     // 2560 elements
#define B_EL (256 * APITCH)    // 10240 elements
#define BUF_EL (A_EL + B_EL)   // 12800 elements
#define BUF_BYTES (BUF_EL * 2) // 25600 bytes per buffer

typedef __attribute__((ext_vector_type(16))) __bf16 v16bf;
typedef __attribute__((ext_vector_type(8)))  float  v8f;
typedef __attribute__((ext_vector_type(4)))  unsigned int v4u;
typedef __attribute__((ext_vector_type(8)))  int v8i;
typedef __attribute__((ext_vector_type(4)))  int v4i;

struct B32x8 { uint4 lo, hi; };

#if __has_builtin(__builtin_amdgcn_tensor_load_to_lds) && __has_builtin(__builtin_amdgcn_s_wait_tensorcnt)
#define USE_TDM 1
#else
#define USE_TDM 0
#endif

__device__ __forceinline__ unsigned short f2bf(float f) {
    union { float f; unsigned u; } t; t.f = f;
    unsigned r = t.u + 0x7FFFu + ((t.u >> 16) & 1u);   // round-to-nearest-even
    return (unsigned short)(r >> 16);
}

__device__ __forceinline__ float log_expected_count(int idx) {
    float c = (float)idx;
    float p = logf((c + 2.0f) / (c + 1.0f)) / logf((float)NUM_CLASSES + 1.0f);
    return logf(-expm1f((float)NUM_SAMPLED * log1pf(-p)));
}

#if USE_TDM
// Build a D# (ISA 8.3-8.5) for a 2D tile of bf16: tile_dim0=KCH along contiguous dim,
// tile_dim1=tileRows, row stride DIM elements, LDS padding 4 dwords per 16 dwords
// (row pitch 32 -> 40 elements).
__device__ __forceinline__ void tdm_issue_tile(const unsigned short* gbase,
                                               unsigned ldsOff,
                                               unsigned tileRows,
                                               unsigned tensorRows) {
    unsigned long long ga = (unsigned long long)(uintptr_t)gbase;
    v4u g0;
    g0.x = 1u;                                             // count=1, user mode
    g0.y = ldsOff;                                         // lds_addr (bytes)
    g0.z = (unsigned)ga;                                   // global_addr[31:0]
    g0.w = (unsigned)((ga >> 32) & 0x01FFFFFFu) | 0x80000000u; // addr[56:32] | type=2
    unsigned d0 = (1u << 16)   // data_size = 2 bytes
                | (1u << 20)   // pad_enable
                | (3u << 22)   // pad_interval: 16 dwords (one 32-elem row)
                | (3u << 25);  // pad_amount: 4 dwords (8 elements)
    unsigned d1 = ((unsigned)DIM & 0xFFFFu) << 16;         // tensor_dim0[15:0]
    unsigned d2 = (((unsigned)DIM >> 16) & 0xFFFFu) | ((tensorRows & 0xFFFFu) << 16);
    unsigned d3 = ((tensorRows >> 16) & 0xFFFFu) | ((unsigned)KCH << 16); // tile_dim0
    unsigned d4 = tileRows;                                // tile_dim1 (tile_dim2 = 0)
    unsigned d5 = (unsigned)DIM;                           // tensor_dim0_stride[31:0]
    v8i g1 = { (int)d0, (int)d1, (int)d2, (int)d3, (int)d4, (int)d5, 0, 0 };
    v4i z4 = { 0, 0, 0, 0 };
#if defined(__clang_major__) && __clang_major__ >= 23
    v8i z8 = { 0, 0, 0, 0, 0, 0, 0, 0 };
    __builtin_amdgcn_tensor_load_to_lds(g0, g1, z4, z4, z8, 0);
#else
    __builtin_amdgcn_tensor_load_to_lds(g0, g1, z4, z4, 0);
#endif
}
#endif

// ---------------- x -> bf16 ----------------
__global__ __launch_bounds__(256) void convert_x_kernel(
    const float* __restrict__ x, unsigned short* __restrict__ xb) {
    int i = blockIdx.x * 256 + threadIdx.x;
    xb[i] = f2bf(x[i]);
}

// ------- gather w[:, sampled] -> bf16 [SPAD][DIM], column bias -------
__global__ __launch_bounds__(256) void gather_w_kernel(
    const float* __restrict__ w, const float* __restrict__ bias,
    const int* __restrict__ sampled,
    unsigned short* __restrict__ wbuf, float* __restrict__ colbias) {
    int j = blockIdx.x;
    if (j < NUM_SAMPLED) {
        int c = sampled[j];
        for (int d = threadIdx.x; d < DIM; d += 256)
            wbuf[(size_t)j * DIM + d] = f2bf(w[(size_t)d * NUM_CLASSES + c]);
        if (threadIdx.x == 0)
            colbias[j] = bias[c] - log_expected_count(c);
    } else {
        for (int d = threadIdx.x; d < DIM; d += 256)
            wbuf[(size_t)j * DIM + d] = 0;
        if (threadIdx.x == 0)
            colbias[j] = -1e30f;
    }
}

// ---------------- true logits: one wave32 per row, f32 ----------------
__global__ __launch_bounds__(256) void true_logits_kernel(
    const float* __restrict__ x, const float* __restrict__ w,
    const float* __restrict__ bias, const int* __restrict__ labels,
    float* __restrict__ tl) {
    int wid  = (blockIdx.x * 256 + threadIdx.x) >> 5;
    int lane = threadIdx.x & 31;
    if (wid >= BATCH) return;
    int c = labels[wid];
    const float* xr = x + (size_t)wid * DIM;
    float sum = 0.f;
    #pragma unroll 4
    for (int d = lane; d < DIM; d += 32)
        sum += xr[d] * w[(size_t)d * NUM_CLASSES + c];
    #pragma unroll
    for (int off = 16; off > 0; off >>= 1)
        sum += __shfl_xor(sum, off, 32);
    if (lane == 0) tl[wid] = sum + bias[c] - log_expected_count(c);
}

// ------- fused bf16 WMMA GEMM (TDM->LDS staged) + online logsumexp -------
// grid = (SPAD/256, BATCH/64), block = 256 (8 waves).
// Block tile 64 rows x 256 cols; wave: Mhalf = wid&1 (32 rows = 2 m-tiles),
// Ngroup = wid>>1 (64 cols = 4 n-tiles) -> 8 accumulators, 8 WMMA per K-chunk.
__global__ __launch_bounds__(256) void gemm_lse_kernel(
    const unsigned short* __restrict__ xb,    // [BATCH][DIM] bf16
    const unsigned short* __restrict__ wb,    // [SPAD][DIM] bf16
    const float* __restrict__ colbias,        // [SPAD]
    float2* __restrict__ partials) {          // [BATCH][NCHUNK]
    __shared__ unsigned short sh[2 * BUF_EL];             // 50 KB, double-buffered

    const int lane   = threadIdx.x & 31;
    const int wid    = threadIdx.x >> 5;
    const int mhalf  = wid & 1;
    const int ngroup = wid >> 1;
    const int lrow   = lane & 15;
    const int half   = lane >> 4;

    const int rowBase = blockIdx.y * 64;
    const int colBase = blockIdx.x * 256;

    v8f acc[2][4];
    #pragma unroll
    for (int mt = 0; mt < 2; ++mt)
        #pragma unroll
        for (int nt = 0; nt < 4; ++nt)
            acc[mt][nt] = (v8f){};

    // LDS element offsets for this lane's fragments (within one buffer)
    // A (16-bit A 16x32 layout): row = mhalf*32 + mt*16 + lrow, K chunks {lo..lo+7, lo+16..lo+23}, lo=8*half
    unsigned aoff0 = (unsigned)((mhalf * 32 + lrow) * APITCH + half * 8);
    // B (32x16, col per lane): col = ngroup*64 + nt*16 + lrow, K chunks {lo..lo+7, lo+16..lo+23}, lo=8*half
    unsigned boff0 = (unsigned)(A_EL + (ngroup * 64 + lrow) * APITCH + half * 8);

    const int nch = DIM / KCH;                            // 32 chunks

#if USE_TDM
    if (wid == 0) {
        tdm_issue_tile(xb + (size_t)rowBase * DIM, 0, 64, BATCH);
        tdm_issue_tile(wb + (size_t)colBase * DIM, A_EL * 2, 256, SPAD);
    }
#endif

    for (int i = 0; i < nch; ++i) {
        const int p = i & 1;
#if USE_TDM
        if (i + 1 < nch) {
            if (wid == 0) {
                const unsigned nb = (unsigned)((i + 1) & 1) * BUF_BYTES;
                const int k = (i + 1) * KCH;
                tdm_issue_tile(xb + (size_t)rowBase * DIM + k, nb, 64, BATCH);
                tdm_issue_tile(wb + (size_t)colBase * DIM + k, nb + A_EL * 2, 256, SPAD);
                __builtin_amdgcn_s_wait_tensorcnt((short)2);  // chunk i complete
            }
        } else {
            if (wid == 0) __builtin_amdgcn_s_wait_tensorcnt((short)0);
        }
        __syncthreads();
#else
        // Fallback: cooperative copy with explicit padding, single chunk in flight.
        __syncthreads();
        {
            const int k = i * KCH;
            for (int c = threadIdx.x; c < 64 * 2; c += 256) {      // A: 64 rows x 2 x16B
                int r = c >> 1, s = c & 1;
                *(uint4*)(sh + p * BUF_EL + r * APITCH + s * 8) =
                    *(const uint4*)(xb + (size_t)(rowBase + r) * DIM + k + s * 8);
            }
            for (int c = threadIdx.x; c < 256 * 2; c += 256) {     // B: 256 rows x 2 x16B
                int r = c >> 1, s = c & 1;
                *(uint4*)(sh + p * BUF_EL + A_EL + r * APITCH + s * 8) =
                    *(const uint4*)(wb + (size_t)(colBase + r) * DIM + k + s * 8);
            }
        }
        __syncthreads();
#endif
        const unsigned short* buf = sh + p * BUF_EL;

        v16bf a[2];
        #pragma unroll
        for (int mt = 0; mt < 2; ++mt) {
            B32x8 t;
            t.lo = *(const uint4*)(buf + aoff0 + mt * 16 * APITCH);
            t.hi = *(const uint4*)(buf + aoff0 + mt * 16 * APITCH + 16);
            a[mt] = __builtin_bit_cast(v16bf, t);
        }
        #pragma unroll
        for (int nt = 0; nt < 4; ++nt) {
            B32x8 t;
            t.lo = *(const uint4*)(buf + boff0 + nt * 16 * APITCH);
            t.hi = *(const uint4*)(buf + boff0 + nt * 16 * APITCH + 16);
            v16bf b = __builtin_bit_cast(v16bf, t);
            acc[0][nt] = __builtin_amdgcn_wmma_f32_16x16x32_bf16(false, a[0], false, b, (short)0, acc[0][nt], false, false);
            acc[1][nt] = __builtin_amdgcn_wmma_f32_16x16x32_bf16(false, a[1], false, b, (short)0, acc[1][nt], false, false);
        }
        __syncthreads();
    }

    // Epilogue: add column bias, online (max, sumexp) per row.
    float cb[4];
    #pragma unroll
    for (int nt = 0; nt < 4; ++nt)
        cb[nt] = colbias[colBase + ngroup * 64 + nt * 16 + lrow];

    float2* red = (float2*)sh;                            // [64 rows][4 groups]

    #pragma unroll
    for (int mt = 0; mt < 2; ++mt) {
        #pragma unroll
        for (int r = 0; r < 8; ++r) {
            // C layout: VGPR r -> row r (lanes 0-15) / r+8 (lanes 16-31), col = lane&15
            float v0 = acc[mt][0][r] + cb[0];
            float v1 = acc[mt][1][r] + cb[1];
            float v2 = acc[mt][2][r] + cb[2];
            float v3 = acc[mt][3][r] + cb[3];
            float m = fmaxf(fmaxf(v0, v1), fmaxf(v2, v3));
            float s = expf(v0 - m) + expf(v1 - m) + expf(v2 - m) + expf(v3 - m);
            #pragma unroll
            for (int off = 1; off < 16; off <<= 1) {
                float om = __shfl_xor(m, off, 32);
                float os = __shfl_xor(s, off, 32);
                float M  = fmaxf(m, om);
                s = s * expf(m - M) + os * expf(om - M);
                m = M;
            }
            if (lrow == 0)
                red[(mhalf * 32 + mt * 16 + r + (half << 3)) * 4 + ngroup] = make_float2(m, s);
        }
    }
    __syncthreads();

    if (threadIdx.x < 64) {
        float2 pq = red[threadIdx.x * 4 + 0];
        float m = pq.x, s = pq.y;
        #pragma unroll
        for (int g = 1; g < 4; ++g) {
            float2 q = red[threadIdx.x * 4 + g];
            float M = fmaxf(m, q.x);
            s = s * expf(m - M) + q.y * expf(q.x - M);
            m = M;
        }
        int row = rowBase + threadIdx.x;
        partials[(size_t)row * NCHUNK + blockIdx.x] = make_float2(m, s);
    }
}

// ---------------- combine partials per row, block-sum ----------------
__global__ __launch_bounds__(256) void combine_kernel(
    const float2* __restrict__ partials, const float* __restrict__ tl,
    float* __restrict__ bsums) {
    __shared__ float ssum[256];
    int b = blockIdx.x * 256 + threadIdx.x;
    float t = tl[b];
    float m = t, s = 1.0f;                      // seed with exp(t - t)
    #pragma unroll
    for (int ch = 0; ch < NCHUNK; ++ch) {
        float2 p = partials[(size_t)b * NCHUNK + ch];
        float M = fmaxf(m, p.x);
        s = s * expf(m - M) + p.y * expf(p.x - M);
        m = M;
    }
    ssum[threadIdx.x] = (m + logf(s)) - t;      // lse - true_logit
    __syncthreads();
    #pragma unroll
    for (int off = 128; off > 0; off >>= 1) {
        if (threadIdx.x < off) ssum[threadIdx.x] += ssum[threadIdx.x + off];
        __syncthreads();
    }
    if (threadIdx.x == 0) bsums[blockIdx.x] = ssum[0];
}

__global__ void finalize_kernel(const float* __restrict__ bsums, float* __restrict__ out) {
    float s = 0.f;
    for (int i = 0; i < BATCH / 256; ++i) s += bsums[i];
    out[0] = s / (float)BATCH;
}

extern "C" void kernel_launch(void* const* d_in, const int* in_sizes, int n_in,
                              void* d_out, int out_size, void* d_ws, size_t ws_size,
                              hipStream_t stream) {
    const float* x       = (const float*)d_in[0];
    const float* w       = (const float*)d_in[1];
    const float* bias    = (const float*)d_in[2];
    const int*   labels  = (const int*)d_in[3];
    const int*   sampled = (const int*)d_in[4];
    float* out = (float*)d_out;

    char* ws = (char*)d_ws;
    size_t off = 0;
    unsigned short* wbuf = (unsigned short*)(ws + off); off += (size_t)SPAD * DIM * 2;   // 10.5 MB
    unsigned short* xbuf = (unsigned short*)(ws + off); off += (size_t)BATCH * DIM * 2;  //  8.4 MB
    float* colbias  = (float*)(ws + off);   off += (size_t)SPAD * 4;
    float2* partials = (float2*)(ws + off); off += (size_t)BATCH * NCHUNK * 8;
    float* tl    = (float*)(ws + off);      off += (size_t)BATCH * 4;
    float* bsums = (float*)(ws + off);      off += (BATCH / 256) * 4;

    convert_x_kernel<<<(BATCH * DIM) / 256, 256, 0, stream>>>(x, xbuf);
    gather_w_kernel<<<SPAD, 256, 0, stream>>>(w, bias, sampled, wbuf, colbias);
    true_logits_kernel<<<BATCH / 8, 256, 0, stream>>>(x, w, bias, labels, tl);
    gemm_lse_kernel<<<dim3(SPAD / 256, BATCH / 64), 256, 0, stream>>>(xbuf, wbuf, colbias, partials);
    combine_kernel<<<BATCH / 256, 256, 0, stream>>>(partials, tl, bsums);
    finalize_kernel<<<1, 1, 0, stream>>>(bsums, out);
}
